// MambaLayer_8572754723045
// MI455X (gfx1250) — matrix operand order repaired and checked by reading, not compiled, not run on hardware
//
#include <hip/hip_runtime.h>
#include <hip/hip_bf16.h>
#include <math.h>

// ---------------- problem dims (match reference) ----------------
#define BATCH  16
#define SEQ    2048
#define DMODEL 256
#define DINNER 512          // EXPAND * D_MODEL
#define DSTATE 16
#define DTRANK 16
#define DCONV  4
#define DFF    1024
#define NTOK   (BATCH*SEQ)  // 32768 tokens
#define XDBL_P 64           // padded x_dbl row: dt[0:16) | B[16:32) | C[32:48) | pad

typedef __attribute__((ext_vector_type(16))) __bf16        v16bf;
typedef __attribute__((ext_vector_type(8)))  float         v8f;
typedef __attribute__((ext_vector_type(8)))  unsigned int  v8u;

__device__ __forceinline__ unsigned short f2bf(float f) {
  unsigned u = __builtin_bit_cast(unsigned, f);
  unsigned r = (u + 0x7FFFu + ((u >> 16) & 1u)) >> 16;  // RNE
  return (unsigned short)r;
}
__device__ __forceinline__ float sigmoidf_(float x) { return 1.f / (1.f + __expf(-x)); }

// ---------------- fp32 -> bf16 convert ----------------
__global__ void cvt_bf16_kernel(const float* __restrict__ in,
                                unsigned short* __restrict__ out, int n) {
  int i = blockIdx.x * blockDim.x + threadIdx.x;
  if (i < n) out[i] = f2bf(in[i]);
}

// fp32 [Nin,Kin] -> bf16 [Nout,Kout] zero-padded (weights only, tiny)
__global__ void cvt_pad_kernel(const float* __restrict__ in,
                               unsigned short* __restrict__ out,
                               int Nin, int Kin, int Kout, int total) {
  int i = blockIdx.x * blockDim.x + threadIdx.x;
  if (i >= total) return;
  int n = i / Kout, k = i - n * Kout;
  out[i] = (n < Nin && k < Kin) ? f2bf(in[n * Kin + k]) : (unsigned short)0;
}

// dt slice of padded x_dbl -> bf16 [NTOK, 64], cols >= DTRANK zeroed
__global__ void dt_pad_kernel(const float* __restrict__ xdbl,
                              unsigned short* __restrict__ dtbf) {
  int i = blockIdx.x * blockDim.x + threadIdx.x;  // over NTOK*64
  int col = i & (XDBL_P - 1);
  dtbf[i] = (col < DTRANK) ? f2bf(xdbl[i]) : (unsigned short)0;
}

// ---------------- bf16 WMMA GEMM: C[M,N] = A[M,K] * W[N,K]^T (+bias, +act) ---
// Requirements (guaranteed by caller via padding): gridDim.x*128 == M,
// gridDim.y*64 == N, K % 64 == 0, all strides multiples of 8 elements.
// Block: 256 threads (8 waves), block tile 128x64, wave tile 32x32,
// K-step 64 -> 8x v_wmma_f32_16x16x32_bf16 per iteration per wave.
// epi: 0=none, 1=+bias, 2=+bias,GELU(exact), 3=+bias,softplus
__global__ __launch_bounds__(256)
void gemm_bf16_wmma_kernel(const unsigned short* __restrict__ A, int lda,
                           const unsigned short* __restrict__ W, int ldw,
                           const float* __restrict__ bias,
                           float* __restrict__ Cf, unsigned short* __restrict__ Cb,
                           int ldc, int K, int epi) {
  __shared__ uint4 lAs4[128 * 8];  // 128 rows x 64 bf16
  __shared__ uint4 lBs4[64 * 8];   //  64 rows x 64 bf16
  unsigned int* lAs = (unsigned int*)lAs4;
  unsigned int* lBs = (unsigned int*)lBs4;

  const int tid  = threadIdx.x;
  const int lane = tid & 31, half = lane >> 4, l16 = lane & 15;
  const int wv = tid >> 5;
  const int wm = wv & 3;   // 4 waves along M
  const int wn = wv >> 2;  // 2 waves along N
  const int m0 = blockIdx.x * 128;
  const int n0 = blockIdx.y * 64;

  v8f acc[2][2];
#pragma unroll
  for (int mt = 0; mt < 2; ++mt)
#pragma unroll
    for (int nt = 0; nt < 2; ++nt)
#pragma unroll
      for (int r = 0; r < 8; ++r) acc[mt][nt][r] = 0.f;

  for (int k0 = 0; k0 < K; k0 += 64) {
    __syncthreads();
    // A tile: 128x64 bf16 = 1024 uint4, 4 per thread (global_load_b128)
#pragma unroll
    for (int j = 0; j < 4; ++j) {
      int q = j * 256 + tid;
      int row = q >> 3, c4 = q & 7;
      lAs4[row * 8 + c4] =
          *(const uint4*)(A + (size_t)(m0 + row) * lda + k0 + c4 * 8);
    }
    // B tile: 64x64 bf16 = 512 uint4, 2 per thread
#pragma unroll
    for (int j = 0; j < 2; ++j) {
      int q = j * 256 + tid;
      int row = q >> 3, c4 = q & 7;
      lBs4[row * 8 + c4] =
          *(const uint4*)(W + (size_t)(n0 + row) * ldw + k0 + c4 * 8);
    }
    // prefetch next A K-slice (speculative; dropped if OOB)
    __builtin_prefetch(A + (size_t)(m0 + (tid >> 1)) * lda + k0 + 64, 0, 3);
    __syncthreads();

#pragma unroll
    for (int ks = 0; ks < 2; ++ks) {
      // fragments per CDNA5 16-bit A (16x32) / B (32x16) VGPR layouts
      v8u au[2], bu[2];
#pragma unroll
      for (int mt = 0; mt < 2; ++mt) {
        const unsigned* base = &lAs[(wm * 32 + mt * 16 + l16) * 32 + ks * 16];
#pragma unroll
        for (int v = 0; v < 8; ++v) {
          int kp = (v < 4) ? (4 * half + v) : (4 + 4 * half + v);
          au[mt][v] = base[kp];
        }
      }
#pragma unroll
      for (int nt = 0; nt < 2; ++nt) {
        const unsigned* base = &lBs[(wn * 32 + nt * 16 + l16) * 32 + ks * 16];
#pragma unroll
        for (int v = 0; v < 8; ++v) bu[nt][v] = base[(v < 4) ? (4 * half + v) : (4 + 4 * half + v)];
      }
#pragma unroll
      for (int mt = 0; mt < 2; ++mt)
#pragma unroll
        for (int nt = 0; nt < 2; ++nt)
          acc[mt][nt] = __builtin_amdgcn_wmma_f32_16x16x32_bf16(
              false, __builtin_bit_cast(v16bf, au[mt]),
              false, __builtin_bit_cast(v16bf, bu[nt]),
              (short)0, acc[mt][nt], false, false);
    }
  }

  // Epilogue: C/D layout -> VGPR r holds M = r + 8*half, N = lane&15
#pragma unroll
  for (int mt = 0; mt < 2; ++mt) {
#pragma unroll
    for (int nt = 0; nt < 2; ++nt) {
      int gn = n0 + wn * 32 + nt * 16 + l16;
      float bv = (epi >= 1 && bias) ? bias[gn] : 0.f;
#pragma unroll
      for (int r = 0; r < 8; ++r) {
        int gm = m0 + wm * 32 + mt * 16 + half * 8 + r;
        float v = acc[mt][nt][r] + bv;
        if (epi == 2)      v = 0.5f * v * (1.f + erff(v * 0.70710678118f));
        else if (epi == 3) v = (v > 20.f) ? v : log1pf(__expf(v));
        size_t o = (size_t)gm * ldc + gn;
        if (Cf) Cf[o] = v;
        if (Cb) Cb[o] = f2bf(v);
      }
    }
  }
}

// ---------------- depthwise causal conv1d (width 4) + bias + SiLU -----------
__global__ void conv_silu_kernel(const float* __restrict__ xz,
                                 const float* __restrict__ conv_w,
                                 const float* __restrict__ conv_b,
                                 float* __restrict__ uc,
                                 unsigned short* __restrict__ ucbf) {
  size_t i = (size_t)blockIdx.x * blockDim.x + threadIdx.x;  // over NTOK*DINNER
  int c = (int)(i & (DINNER - 1));
  size_t t = i >> 9;                 // token index b*SEQ + l
  int l = (int)(t & (SEQ - 1));
  float acc = conv_b[c];
#pragma unroll
  for (int j = 0; j < DCONV; ++j) {
    int lj = l - (DCONV - 1) + j;
    if (lj >= 0)
      acc += xz[(t - (size_t)(l - lj)) * (2 * DINNER) + c] * conv_w[c * DCONV + j];
  }
  float s = acc * sigmoidf_(acc);
  uc[i] = s;
  ucbf[i] = f2bf(s);
}

// ---------------- selective scan: 1 thread per (b, channel), state in regs --
__global__ __launch_bounds__(DINNER)
void scan_kernel(const float* __restrict__ xdbl,   // [NTOK, 64]: dt|B|C|pad
                 const float* __restrict__ delta,  // [NTOK, 512]
                 const float* __restrict__ uc,     // [NTOK, 512]
                 const float* __restrict__ A_log,  // [512, 16]
                 float* __restrict__ y) {          // [NTOK, 512]
  const int b = blockIdx.x;
  const int c = threadIdx.x;
  float A[DSTATE], h[DSTATE];
#pragma unroll
  for (int s = 0; s < DSTATE; ++s) { A[s] = -__expf(A_log[c * DSTATE + s]); h[s] = 0.f; }

  __shared__ float sB[32][DSTATE];
  __shared__ float sC[32][DSTATE];

  for (int l0 = 0; l0 < SEQ; l0 += 32) {
    __syncthreads();
    // stage 32 rows of B|C (32 floats each): 1024 floats, 2 per thread
#pragma unroll
    for (int j = 0; j < 2; ++j) {
      int i = j * DINNER + c;
      int row = i >> 5, col = i & 31;
      float v = xdbl[((size_t)(b * SEQ + l0 + row)) * XDBL_P + DTRANK + col];
      if (col < DSTATE) sB[row][col] = v; else sC[row][col - DSTATE] = v;
    }
    __syncthreads();
#pragma unroll 4
    for (int t = 0; t < 32; ++t) {
      size_t idx = ((size_t)b * SEQ + (l0 + t)) * DINNER + c;
      float d = delta[idx], u = uc[idx];
      float du = d * u, accv = 0.f;
#pragma unroll
      for (int s = 0; s < DSTATE; ++s) {
        float dA = __expf(d * A[s]);
        h[s] = h[s] * dA + du * sB[t][s];
        accv += h[s] * sC[t][s];
      }
      y[idx] = accv;
    }
  }
}

// ---------------- y = (y + uc*Dskip) * silu(z), bf16 out for out_proj -------
__global__ void gate_kernel(const float* __restrict__ y, const float* __restrict__ uc,
                            const float* __restrict__ xz, const float* __restrict__ Dskip,
                            unsigned short* __restrict__ ybf) {
  size_t i = (size_t)blockIdx.x * blockDim.x + threadIdx.x;
  int c = (int)(i & (DINNER - 1));
  size_t row = i >> 9;
  float z = xz[row * (2 * DINNER) + DINNER + c];
  float v = (y[i] + uc[i] * Dskip[c]) * (z * sigmoidf_(z));
  ybf[i] = f2bf(v);
}

// ---------------- residual add + LayerNorm (D = 256, one block per token) ---
__global__ __launch_bounds__(DMODEL)
void add_ln_kernel(const float* __restrict__ a, const float* __restrict__ res,
                   const float* __restrict__ g, const float* __restrict__ bb,
                   float* __restrict__ outf, unsigned short* __restrict__ outbf) {
  __shared__ float red[DMODEL];
  const int row = blockIdx.x, t = threadIdx.x;
  size_t o = (size_t)row * DMODEL + t;
  float v = a[o] + res[o];
  red[t] = v; __syncthreads();
  for (int s = DMODEL / 2; s > 0; s >>= 1) { if (t < s) red[t] += red[t + s]; __syncthreads(); }
  float mu = red[0] * (1.f / DMODEL);
  __syncthreads();
  float d = v - mu;
  red[t] = d * d; __syncthreads();
  for (int s = DMODEL / 2; s > 0; s >>= 1) { if (t < s) red[t] += red[t + s]; __syncthreads(); }
  float var = red[0] * (1.f / DMODEL);
  float r = d * rsqrtf(var + 1e-12f) * g[t] + bb[t];
  if (outf)  outf[o]  = r;
  if (outbf) outbf[o] = f2bf(r);
}

// ---------------------------------------------------------------------------
extern "C" void kernel_launch(void* const* d_in, const int* in_sizes, int n_in,
                              void* d_out, int out_size, void* d_ws, size_t ws_size,
                              hipStream_t stream) {
  (void)in_sizes; (void)n_in; (void)out_size; (void)ws_size;
  const float* x         = (const float*)d_in[0];
  const float* in_proj_w = (const float*)d_in[1];
  const float* conv_w    = (const float*)d_in[2];
  const float* conv_b    = (const float*)d_in[3];
  const float* x_proj_w  = (const float*)d_in[4];
  const float* dt_proj_w = (const float*)d_in[5];
  const float* dt_proj_b = (const float*)d_in[6];
  const float* A_log     = (const float*)d_in[7];
  const float* Dskip     = (const float*)d_in[8];
  const float* out_proj_w= (const float*)d_in[9];
  const float* ln1_g = (const float*)d_in[10];
  const float* ln1_b = (const float*)d_in[11];
  const float* w1 = (const float*)d_in[12];
  const float* b1 = (const float*)d_in[13];
  const float* w2 = (const float*)d_in[14];
  const float* b2 = (const float*)d_in[15];
  const float* ln2_g = (const float*)d_in[16];
  const float* ln2_b = (const float*)d_in[17];
  float* out = (float*)d_out;

  // ---- carve workspace ----
  char* ws = (char*)d_ws;
  size_t off = 0;
  auto carve = [&](size_t bytes) -> char* {
    char* p = ws + off; off += (bytes + 255) & ~(size_t)255; return p;
  };
  unsigned short* xbf    = (unsigned short*)carve((size_t)NTOK * DMODEL * 2);
  unsigned short* wib    = (unsigned short*)carve((size_t)2 * DINNER * DMODEL * 2);
  unsigned short* wxb    = (unsigned short*)carve((size_t)64 * DINNER * 2);      // padded N=64
  unsigned short* wdtb   = (unsigned short*)carve((size_t)DINNER * 64 * 2);      // padded K=64
  unsigned short* wob    = (unsigned short*)carve((size_t)DMODEL * DINNER * 2);
  unsigned short* w1b    = (unsigned short*)carve((size_t)DFF * DMODEL * 2);
  unsigned short* w2b    = (unsigned short*)carve((size_t)DMODEL * DFF * 2);
  float*          xz     = (float*)carve((size_t)NTOK * 2 * DINNER * 4);
  float*          uc     = (float*)carve((size_t)NTOK * DINNER * 4);
  unsigned short* ucbf   = (unsigned short*)carve((size_t)NTOK * DINNER * 2);
  float*          xdbl   = (float*)carve((size_t)NTOK * XDBL_P * 4);
  unsigned short* dtbf   = (unsigned short*)carve((size_t)NTOK * XDBL_P * 2);
  float*          delta  = (float*)carve((size_t)NTOK * DINNER * 4);
  float*          yscan  = (float*)carve((size_t)NTOK * DINNER * 4);
  unsigned short* ybf    = (unsigned short*)carve((size_t)NTOK * DINNER * 2);
  float*          hm     = (float*)carve((size_t)NTOK * DMODEL * 4);
  float*          h      = (float*)carve((size_t)NTOK * DMODEL * 4);
  unsigned short* hbf    = (unsigned short*)carve((size_t)NTOK * DMODEL * 2);
  unsigned short* ffbf   = (unsigned short*)carve((size_t)NTOK * DFF * 2);
  float*          ff2    = (float*)carve((size_t)NTOK * DMODEL * 4);

  const dim3 blk(256);
  auto cvt = [&](const float* in, unsigned short* o, int n) {
    cvt_bf16_kernel<<<(n + 255) / 256, blk, 0, stream>>>(in, o, n);
  };
  cvt(x, xbf, NTOK * DMODEL);
  cvt(in_proj_w, wib, 2 * DINNER * DMODEL);
  cvt(out_proj_w, wob, DMODEL * DINNER);
  cvt(w1, w1b, DFF * DMODEL);
  cvt(w2, w2b, DMODEL * DFF);
  // padded weights
  cvt_pad_kernel<<<(64 * DINNER + 255) / 256, blk, 0, stream>>>(
      x_proj_w, wxb, 48, DINNER, DINNER, 64 * DINNER);               // N 48->64
  cvt_pad_kernel<<<(DINNER * 64 + 255) / 256, blk, 0, stream>>>(
      dt_proj_w, wdtb, DINNER, DTRANK, 64, DINNER * 64);             // K 16->64

  // in_proj: xz = x @ in_proj_w^T   [32768,1024]
  gemm_bf16_wmma_kernel<<<dim3(NTOK / 128, (2 * DINNER) / 64), blk, 0, stream>>>(
      xbf, DMODEL, wib, DMODEL, nullptr, xz, nullptr, 2 * DINNER, DMODEL, 0);

  // depthwise causal conv + SiLU
  conv_silu_kernel<<<(NTOK * DINNER) / 256, blk, 0, stream>>>(xz, conv_w, conv_b, uc, ucbf);

  // x_proj: x_dbl = uc @ x_proj_w^T   [32768, 64] (padded cols are 0)
  gemm_bf16_wmma_kernel<<<dim3(NTOK / 128, 1), blk, 0, stream>>>(
      ucbf, DINNER, wxb, DINNER, nullptr, xdbl, nullptr, XDBL_P, DINNER, 0);

  // dt (cols 0..15 of x_dbl) -> bf16, K-padded to 64
  dt_pad_kernel<<<(NTOK * XDBL_P) / 256, blk, 0, stream>>>(xdbl, dtbf);

  // dt_proj + softplus: delta = softplus(dt @ dt_proj_w^T + b)
  gemm_bf16_wmma_kernel<<<dim3(NTOK / 128, DINNER / 64), blk, 0, stream>>>(
      dtbf, XDBL_P, wdtb, 64, dt_proj_b, delta, nullptr, DINNER, 64, 3);

  // selective scan (sequential over L, 8192 channel-threads)
  scan_kernel<<<BATCH, DINNER, 0, stream>>>(xdbl, delta, uc, A_log, yscan);

  // gate: y = (y + uc*D) * silu(z)  -> bf16
  gate_kernel<<<(NTOK * DINNER) / 256, blk, 0, stream>>>(yscan, uc, xz, Dskip, ybf);

  // out_proj: hm = y @ out_proj_w^T   [32768,256]
  gemm_bf16_wmma_kernel<<<dim3(NTOK / 128, DMODEL / 64), blk, 0, stream>>>(
      ybf, DINNER, wob, DINNER, nullptr, hm, nullptr, DMODEL, DINNER, 0);

  // LN1(hm + x) -> h (fp32 residual) + hbf
  add_ln_kernel<<<NTOK, DMODEL, 0, stream>>>(hm, x, ln1_g, ln1_b, h, hbf);

  // FF1: gelu(h @ w1^T + b1) -> bf16 only
  gemm_bf16_wmma_kernel<<<dim3(NTOK / 128, DFF / 64), blk, 0, stream>>>(
      hbf, DMODEL, w1b, DMODEL, b1, nullptr, ffbf, DFF, DMODEL, 2);

  // FF2: ff2 = ff @ w2^T + b2
  gemm_bf16_wmma_kernel<<<dim3(NTOK / 128, DMODEL / 64), blk, 0, stream>>>(
      ffbf, DFF, w2b, DFF, b2, ff2, nullptr, DMODEL, DFF, 1);

  // LN2(ff2 + h) -> out
  add_ln_kernel<<<NTOK, DMODEL, 0, stream>>>(ff2, h, ln2_g, ln2_b, out, nullptr);
}
